// NEA_GNN_75608604279024
// MI455X (gfx1250) — matrix operand order repaired and checked by reading, not compiled, not run on hardware
//
#include <hip/hip_runtime.h>
#include <hip/hip_bf16.h>
#include <stdint.h>

typedef __attribute__((ext_vector_type(16))) __bf16 v16bf;
typedef __attribute__((ext_vector_type(8)))  float  v8f;

__device__ __forceinline__ __bf16 f2bf(float x) { return (__bf16)x; }

// ---------------- utility kernels ----------------

__global__ void k_zero(float* __restrict__ p, long n) {
  long i = (long)blockIdx.x * blockDim.x + threadIdx.x;
  if (i < n) p[i] = 0.0f;
}

__global__ void k_count_deg(const int* __restrict__ ne_node, float* __restrict__ deg, int twoE) {
  int t = blockIdx.x * blockDim.x + threadIdx.x;
  if (t < twoE) atomicAdd(&deg[ne_node[t]], 1.0f);
}

__global__ void k_invert_deg(float* __restrict__ deg, int n) {
  int i = blockIdx.x * blockDim.x + threadIdx.x;
  if (i < n) deg[i] = 1.0f / fmaxf(deg[i], 1.0f);
}

// scatter-add x_e rows into per-node accumulator (4 floats per thread)
__global__ void k_scatter_e2v(const int* __restrict__ nei, const float* __restrict__ xe,
                              float* __restrict__ acc, int twoE, int C) {
  int idx = blockIdx.x * blockDim.x + threadIdx.x;
  int c4count = C >> 2;
  int t = idx / c4count;
  if (t >= twoE) return;
  int c = (idx - t * c4count) << 2;
  int node = nei[t];
  int edge = nei[twoE + t];
  const float* s = xe + (size_t)edge * C + c;
  float* d = acc + (size_t)node * C + c;
  atomicAdd(d + 0, s[0]);
  atomicAdd(d + 1, s[1]);
  atomicAdd(d + 2, s[2]);
  atomicAdd(d + 3, s[3]);
}

// transpose f32 weight [K x 64] row-major -> bf16 [64 x K] column-contiguous
__global__ void k_wt_prep(const float* __restrict__ W, __bf16* __restrict__ Wt, int K) {
  int idx = blockIdx.x * blockDim.x + threadIdx.x;
  if (idx >= K * 64) return;
  int k = idx >> 6, n = idx & 63;
  Wt[(size_t)n * K + k] = f2bf(W[(size_t)k * 64 + n]);
}

// ---------------- WMMA dual-GEMM ----------------
// Computes, per 16-row tile (one wave):
//   P1 = A1[tile] @ W1   (K1 x 64)
//   P2 = A2'[tile] @ W2  (K2 x 64)  where A2' is either plain*scale or 0.5*(xv[src]+xv[dst])
// CONCAT: C[:,0:64]=relu(P1), C[:,64:128]=relu(P2);  else C[:,0:64]=relu(P1+P2)

__device__ __forceinline__ v16bf load_b_frag(const __bf16* __restrict__ Wt, int K,
                                             int col, int k, int half) {
  typedef __attribute__((ext_vector_type(8))) __bf16 v8bf;
  const __bf16* p = Wt + (size_t)col * K + k + half * 8;
  v8bf lo = *(const v8bf*)(p);
  v8bf hi = *(const v8bf*)(p + 16);
  return __builtin_shufflevector(lo, hi, 0, 1, 2, 3, 4, 5, 6, 7, 8, 9, 10, 11, 12, 13, 14, 15);
}

template<bool CONCAT, bool GATHER>
__global__ __launch_bounds__(128)
void k_gemm_dual(const float* __restrict__ A1, int K1, const __bf16* __restrict__ W1,
                 const float* __restrict__ A2, int K2, const __bf16* __restrict__ W2,
                 const float* __restrict__ scale2,
                 const int* __restrict__ eidx, int nE,
                 float* __restrict__ C, int M, int ldc) {
  const int lane = threadIdx.x & 31;
  const int wid  = threadIdx.x >> 5;
  const int tile = blockIdx.x * 4 + wid;
  const int row0 = tile * 16;
  if (row0 >= M) return;                // wave-uniform: EXEC stays all-ones for WMMA
  const int l15  = lane & 15;
  const int half = lane >> 4;           // A-frag: half0 -> K {0..7,16..23}, half1 -> {8..15,24..31}
  const int arow = row0 + l15;

  v8f acc1[4] = {};
  v8f acc2[4] = {};

  // ---- operand 1: plain A1 @ W1 ----
  {
    const float* base = A1 + (size_t)arow * K1 + half * 8;
    for (int k = 0; k < K1; k += 32) {
      v16bf a;
#pragma unroll
      for (int i = 0; i < 8; i++) a[i]     = f2bf(base[k + i]);
#pragma unroll
      for (int i = 0; i < 8; i++) a[8 + i] = f2bf(base[k + 16 + i]);
#pragma unroll
      for (int j = 0; j < 4; j++) {
        v16bf b = load_b_frag(W1, K1, j * 16 + l15, k, half);
        acc1[j] = __builtin_amdgcn_wmma_f32_16x16x32_bf16(
            false, a, false, b, (short)0, acc1[j], false, false);
      }
    }
  }

  // ---- operand 2: (scaled A2 | gathered endpoint mean) @ W2 ----
  {
    int srcRow = arow, dstRow = arow;
    float s = 1.0f;
    if (GATHER) { srcRow = eidx[arow]; dstRow = eidx[nE + arow]; }
    else if (scale2) { s = scale2[arow]; }
    const float* bs = A2 + (size_t)srcRow * K2 + half * 8;
    const float* bd = A2 + (size_t)dstRow * K2 + half * 8;
    for (int k = 0; k < K2; k += 32) {
      v16bf a;
      if (GATHER) {
#pragma unroll
        for (int i = 0; i < 8; i++) a[i]     = f2bf(0.5f * (bs[k + i] + bd[k + i]));
#pragma unroll
        for (int i = 0; i < 8; i++) a[8 + i] = f2bf(0.5f * (bs[k + 16 + i] + bd[k + 16 + i]));
      } else {
#pragma unroll
        for (int i = 0; i < 8; i++) a[i]     = f2bf(bs[k + i] * s);
#pragma unroll
        for (int i = 0; i < 8; i++) a[8 + i] = f2bf(bs[k + 16 + i] * s);
      }
      v8f* acc = CONCAT ? acc2 : acc1;
#pragma unroll
      for (int j = 0; j < 4; j++) {
        v16bf b = load_b_frag(W2, K2, j * 16 + l15, k, half);
        acc[j] = __builtin_amdgcn_wmma_f32_16x16x32_bf16(
            false, a, false, b, (short)0, acc[j], false, false);
      }
    }
  }

  // ---- store with ReLU; C/D layout: VGPR r -> row (half*8 + r), col = lane&15 ----
  const int orow = row0 + half * 8;
#pragma unroll
  for (int j = 0; j < 4; j++)
#pragma unroll
    for (int r = 0; r < 8; r++)
      C[(size_t)(orow + r) * ldc + j * 16 + l15] = fmaxf(acc1[j][r], 0.0f);
  if (CONCAT) {
#pragma unroll
    for (int j = 0; j < 4; j++)
#pragma unroll
      for (int r = 0; r < 8; r++)
        C[(size_t)(orow + r) * ldc + 64 + j * 16 + l15] = fmaxf(acc2[j][r], 0.0f);
  }
}

// ---------------- host launcher ----------------

extern "C" void kernel_launch(void* const* d_in, const int* in_sizes, int n_in,
                              void* d_out, int out_size, void* d_ws, size_t ws_size,
                              hipStream_t stream) {
  const int N = in_sizes[0] / 32;     // 50000
  const int E = in_sizes[1] / 32;     // 800000
  const int twoE = 2 * E;

  const float* x_v        = (const float*)d_in[0];
  const float* x_e        = (const float*)d_in[1];
  const int*   edge_index = (const int*)d_in[2];   // [2,E]: src | dst
  const int*   nei        = (const int*)d_in[3];   // [2,2E]: node | edge

  char* ws = (char*)d_ws;
  size_t off = 0;
  auto alloc = [&](size_t bytes) -> void* {
    void* p = ws + off;
    off = (off + bytes + 255) & ~(size_t)255;
    return p;
  };
  float*  inv_deg = (float*) alloc((size_t)N * 4);
  float*  e2v     = (float*) alloc((size_t)N * 128 * 4);
  float*  vA      = (float*) alloc((size_t)N * 128 * 4);
  float*  vB      = (float*) alloc((size_t)N * 64 * 4);
  float*  eA      = (float*) alloc((size_t)E * 64 * 4);
  float*  eB      = (float*) alloc((size_t)E * 128 * 4);
  __bf16* wt0     = (__bf16*)alloc((size_t)16 * 128 * 64 * 2);

  // --- bf16 column-major weight prep (16 weights; K per weight per layer shapes) ---
  static const int Kw[16] = {32, 32, 32, 32,   128, 64, 64, 128,
                             64, 128, 128, 64, 128, 64, 64, 128};
  for (int i = 0; i < 16; i++) {
    int K = Kw[i], n = K * 64;
    k_wt_prep<<<(n + 255) / 256, 256, 0, stream>>>(
        (const float*)d_in[4 + i], wt0 + (size_t)i * 128 * 64, K);
  }

  // --- inverse degree ---
  k_zero<<<(N + 255) / 256, 256, 0, stream>>>(inv_deg, N);
  k_count_deg<<<(twoE + 255) / 256, 256, 0, stream>>>(nei, inv_deg, twoE);
  k_invert_deg<<<(N + 255) / 256, 256, 0, stream>>>(inv_deg, N);

  float* xe_final = (float*)d_out;                    // [E,128]
  float* xv_final = (float*)d_out + (size_t)E * 128;  // [N,64]

  const float* xv_cur = x_v; int Cv = 32;
  const float* xe_cur = x_e; int Ce = 32;
  float* vout[4] = {vA, vB, vA, xv_final};
  float* eout[4] = {eA, eB, eA, xe_final};

  for (int l = 0; l < 4; l++) {
    const bool even = (l & 1) == 0;
    const __bf16* Wvr = wt0 + (size_t)(4 * l + 0) * 128 * 64;
    const __bf16* Wva = wt0 + (size_t)(4 * l + 1) * 128 * 64;
    const __bf16* Wer = wt0 + (size_t)(4 * l + 2) * 128 * 64;
    const __bf16* Wea = wt0 + (size_t)(4 * l + 3) * 128 * 64;

    // e2v mean aggregation accumulator
    long accN = (long)N * Ce;
    k_zero<<<(int)((accN + 255) / 256), 256, 0, stream>>>(e2v, accN);
    long scatN = (long)twoE * (Ce >> 2);
    k_scatter_e2v<<<(int)((scatN + 255) / 256), 256, 0, stream>>>(nei, xe_cur, e2v, twoE, Ce);

    const int nblk = (N / 16 + 3) / 4;   // 4 waves (tiles) per 128-thread block
    const int eblk = (E / 16 + 3) / 4;
    if (even) {
      // x_v = relu([x_v@Wvr , (e2v*invdeg)@Wva]);  x_e = relu(x_e@Wer + v2e@Wea)
      k_gemm_dual<true,  false><<<nblk, 128, 0, stream>>>(
          xv_cur, Cv, Wvr, e2v, Ce, Wva, inv_deg, nullptr, 0, vout[l], N, 128);
      k_gemm_dual<false, true ><<<eblk, 128, 0, stream>>>(
          xe_cur, Ce, Wer, xv_cur, Cv, Wea, nullptr, edge_index, E, eout[l], E, 64);
    } else {
      // x_v = relu(x_v@Wvr + (e2v*invdeg)@Wva);  x_e = relu([x_e@Wer , v2e@Wea])
      k_gemm_dual<false, false><<<nblk, 128, 0, stream>>>(
          xv_cur, Cv, Wvr, e2v, Ce, Wva, inv_deg, nullptr, 0, vout[l], N, 64);
      k_gemm_dual<true,  true ><<<eblk, 128, 0, stream>>>(
          xe_cur, Ce, Wer, xv_cur, Cv, Wea, nullptr, edge_index, E, eout[l], E, 128);
    }
    xv_cur = vout[l]; xe_cur = eout[l];
    Cv = even ? 128 : 64;
    Ce = even ? 64 : 128;
  }
  (void)n_in; (void)out_size; (void)ws_size;
}
